// BiasAttention_86371792322641
// MI455X (gfx1250) — matrix operand order repaired and checked by reading, not compiled, not run on hardware
//
#include <hip/hip_runtime.h>

// Problem constants (B, Q, CV, H, CH) = (2, 2048, 512, 8, 64)
#define Bb   2
#define Qq   2048
#define CVc  512
#define Hh   8
#define CHh  64
#define HD   512          // H*CH
#define BQ   4096         // B*Q

typedef __attribute__((ext_vector_type(16))) __bf16 v16bf;
typedef __attribute__((ext_vector_type(8)))  float  v8f;
typedef __attribute__((ext_vector_type(8)))  int    v8i;

union Frag { v8i i; v16bf h; };

__device__ __forceinline__ unsigned short f2bf(float f) {
  unsigned u = __float_as_uint(f);
  u += 0x7FFFu + ((u >> 16) & 1u);           // round-to-nearest-even
  return (unsigned short)(u >> 16);
}
__device__ __forceinline__ int pack_bf(float lo, float hi) {
  return (int)((unsigned)f2bf(lo) | ((unsigned)f2bf(hi) << 16));
}

// Async DMA: global -> LDS, 16 bytes per lane, tracked by ASYNCcnt.
// lds_off = low 32 bits of generic shared address (aperture maps addr[31:0] to LDS).
__device__ __forceinline__ void async_b128(unsigned lds_off, const float* g) {
  asm volatile("global_load_async_to_lds_b128 %0, %1, off"
               :: "v"(lds_off), "v"((unsigned long long)(size_t)g)
               : "memory");
}
__device__ __forceinline__ void wait_async16() {
  asm volatile("s_wait_asynccnt 0x10" ::: "memory");
}
__device__ __forceinline__ void wait_async0() {
  asm volatile("s_wait_asynccnt 0x0" ::: "memory");
}

// ---------------------------------------------------------------------------
// Conversion kernels
// ---------------------------------------------------------------------------
__global__ void k_cvt_x(const float* __restrict__ x, unsigned short* __restrict__ xb, int n) {
  int i = blockIdx.x * blockDim.x + threadIdx.x;
  if (i < n) xb[i] = f2bf(x[i]);
}

// w: [K=512][N=512] row-major fp32  ->  wt: [n][k] bf16 (N-major for B-fragments)
__global__ void k_cvt_wT(const float* __restrict__ w, unsigned short* __restrict__ wt) {
  int k = blockIdx.x * blockDim.x + threadIdx.x;   // 0..511
  int n = blockIdx.y;                              // 0..511
  wt[(size_t)n * 512 + k] = f2bf(w[(size_t)k * 512 + n]);
}

// ---------------------------------------------------------------------------
// One wave computes C[16 x 64] += A[16 x K] * Bt[64 x K]^T   (bf16 in, f32 acc)
// Loads A-fragment + all 4 B-fragments first (issues as one load clause),
// then 4 back-to-back WMMAs -> one wait per group instead of per WMMA.
// ---------------------------------------------------------------------------
__device__ __forceinline__ void gemm_tile(const int* __restrict__ A, int lda,
                                          const int* __restrict__ Bt, int ldb,
                                          int nKints, v8f acc[4]) {
  const int lane = threadIdx.x & 31;
  const int sub  = lane & 15;
  const int half = lane >> 4;
  for (int kc = 0; kc < nKints; kc += 16) {        // 32 bf16 of K per step
    Frag a;
#pragma unroll
    for (int r = 0; r < 4; ++r)                    // VGPR0..3: K = 2r + 8*half
      a.i[r] = A[(size_t)sub * lda + kc + r + 4 * half];
#pragma unroll
    for (int r = 4; r < 8; ++r)                    // VGPR4..7: K = 16 + 2(r-4) + 8*half
      a.i[r] = A[(size_t)sub * lda + kc + 8 + (r - 4) + 4 * half];
    Frag b[4];
#pragma unroll
    for (int t = 0; t < 4; ++t)
#pragma unroll
      for (int r = 0; r < 8; ++r)                  // VGPRr: K = 2r + 8*half (+8 for r>=4)
        b[t].i[r] = Bt[(size_t)(t * 16 + sub) * ldb + kc + r + (r >= 4 ? 4 : 0) + 4 * half];
#pragma unroll
    for (int t = 0; t < 4; ++t)
      acc[t] = __builtin_amdgcn_wmma_f32_16x16x32_bf16(
          false, a.h, false, b[t].h, (short)0, acc[t], false, false);
  }
}

// ---------------------------------------------------------------------------
// v = x@W_v (store transposed bf16 [b,h,d,q]) and g = sigmoid(x@W_g) (fp32)
// grid: (BQ/16, HD/64, 2), block: 32
// ---------------------------------------------------------------------------
__global__ void k_vg(const unsigned short* __restrict__ xb,
                     const unsigned short* __restrict__ WvT,
                     const unsigned short* __restrict__ WgT,
                     unsigned short* __restrict__ vT, float* __restrict__ g) {
  const int mt = blockIdx.x, nt = blockIdx.y, which = blockIdx.z;
  const int lane = threadIdx.x & 31, sub = lane & 15, half = lane >> 4;
  const int row0 = mt * 16, n0 = nt * 64;
  v8f acc[4];
#pragma unroll
  for (int t = 0; t < 4; ++t)
#pragma unroll
    for (int j = 0; j < 8; ++j) acc[t][j] = 0.0f;

  const int* A  = (const int*)xb + (size_t)row0 * (CVc / 2);
  const int* Bt = (const int*)(which ? WgT : WvT) + (size_t)n0 * (CVc / 2);
  gemm_tile(A, CVc / 2, Bt, CVc / 2, CVc / 2, acc);

#pragma unroll
  for (int t = 0; t < 4; ++t)
#pragma unroll
    for (int j = 0; j < 8; ++j) {
      int row = row0 + j + 8 * half;               // b*Q + q
      int col = n0 + t * 16 + sub;                 // h*64 + d
      float val = acc[t][j];
      if (which == 0) {
        int b = row >> 11, q = row & 2047;
        int h = col >> 6,  d = col & 63;
        vT[(((size_t)(b * Hh + h) * CHh + d) << 11) + q] = f2bf(val);
      } else {
        g[(size_t)row * HD + col] = 1.0f / (1.0f + __expf(-val));
      }
    }
}

// ---------------------------------------------------------------------------
// Fused: o = softmax(bias1+bias2) @ v ; og = bf16(o * g)
// One wave per (b, h, 32 q-rows): every lane owns one softmax row.
// bias1/bias2 streamed via async global->LDS b128 with ping-pong buffers.
// grid: (Q/32, H, B), block: 32
// ---------------------------------------------------------------------------
#define XS 36   // LDS row stride (floats): 16B-aligned rows, stride=4 mod 32 banks

__global__ void k_attn(const float* __restrict__ bias1, const float* __restrict__ bias2,
                       const unsigned short* __restrict__ vT, const float* __restrict__ g,
                       unsigned short* __restrict__ og) {
  __shared__ __align__(16) float sm_b1[2][32 * XS];  // bias1 chunk, ping-pong
  __shared__ __align__(16) float sm_b2[2][32 * XS];  // bias2 chunk, ping-pong
  __shared__ int   sm_p[32 * 17];                    // probabilities (bf16 pairs)
  __shared__ float sm_s[32];                         // per-row alpha / 1/l

  const int qt = blockIdx.x, h = blockIdx.y, b = blockIdx.z;
  const int lane = threadIdx.x & 31, sub = lane & 15, half = lane >> 4;
  const int q0 = qt * 32;

  const float* b1  = bias1 + ((size_t)(b * Hh + h) * Qq + q0) * Qq;
  const float* b2  = bias2 + ((size_t)b * Qq + q0) * Qq;
  const int*   vtb = (const int*)vT + (size_t)(b * Hh + h) * CHh * (Qq / 2);

  // async staging geometry: each lane moves 16B; 32 lanes = 4 rows per instr
  const int rg = lane >> 3;          // 0..3 (row within group of 4)
  const int c4 = (lane & 7) * 4;     // column (floats)

  v8f acc[2][4];
#pragma unroll
  for (int a = 0; a < 2; ++a)
#pragma unroll
    for (int t = 0; t < 4; ++t)
#pragma unroll
      for (int j = 0; j < 8; ++j) acc[a][t][j] = 0.0f;

  float m_prev = -3.0e38f, l = 0.0f;

  // prologue: stream chunk 0 into buffer 0 (16 async b128s)
#pragma unroll
  for (int i = 0; i < 32; i += 4) {
    int row = i + rg;
    async_b128((unsigned)(size_t)&sm_b1[0][row * XS + c4], b1 + (size_t)row * Qq + c4);
    async_b128((unsigned)(size_t)&sm_b2[0][row * XS + c4], b2 + (size_t)row * Qq + c4);
  }

  for (int k0 = 0; k0 < Qq; k0 += 32) {
    const int p = (k0 >> 5) & 1;
    const bool more = (k0 + 32) < Qq;

    // stream next chunk into the other buffer while we consume this one
    if (more) {
#pragma unroll
      for (int i = 0; i < 32; i += 4) {
        int row = i + rg;
        async_b128((unsigned)(size_t)&sm_b1[p ^ 1][row * XS + c4],
                   b1 + (size_t)row * Qq + (k0 + 32) + c4);
        async_b128((unsigned)(size_t)&sm_b2[p ^ 1][row * XS + c4],
                   b2 + (size_t)row * Qq + (k0 + 32) + c4);
      }
    }

    // issue B fragments early (L2-resident vT), overlap with softmax below
    Frag bf[4];
#pragma unroll
    for (int t = 0; t < 4; ++t)
#pragma unroll
      for (int r = 0; r < 8; ++r)
        bf[t].i[r] = vtb[(size_t)(t * 16 + sub) * (Qq / 2) + (k0 >> 1) + r + (r >= 4 ? 4 : 0) + 4 * half];

    // wait for THIS chunk's async data (next chunk's 16 may stay in flight)
    if (more) wait_async16(); else wait_async0();

    // online softmax: lane r owns row r (all 32 lanes busy)
    {
      const float4* r1 = (const float4*)&sm_b1[p][lane * XS];
      const float4* r2 = (const float4*)&sm_b2[p][lane * XS];
      float xs[32];
#pragma unroll
      for (int j4 = 0; j4 < 8; ++j4) {
        float4 u = r1[j4], w = r2[j4];
        xs[4 * j4 + 0] = u.x + w.x;
        xs[4 * j4 + 1] = u.y + w.y;
        xs[4 * j4 + 2] = u.z + w.z;
        xs[4 * j4 + 3] = u.w + w.w;
      }
      float mx = -3.0e38f;
#pragma unroll
      for (int j = 0; j < 32; ++j) mx = fmaxf(mx, xs[j]);
      float mn    = fmaxf(m_prev, mx);
      float alpha = __expf(m_prev - mn);
      float s = 0.0f;
#pragma unroll
      for (int j = 0; j < 32; j += 2) {
        float p0 = __expf(xs[j]     - mn);
        float p1 = __expf(xs[j + 1] - mn);
        s += p0 + p1;
        sm_p[lane * 17 + (j >> 1)] = pack_bf(p0, p1);
      }
      l = l * alpha + s;
      m_prev = mn;
      sm_s[lane] = alpha;
    }
    __syncthreads();

    // rescale accumulators by per-row alpha
#pragma unroll
    for (int a = 0; a < 2; ++a) {
      float alp[8];
#pragma unroll
      for (int j = 0; j < 8; ++j) alp[j] = sm_s[a * 16 + j + 8 * half];
#pragma unroll
      for (int t = 0; t < 4; ++t)
#pragma unroll
        for (int j = 0; j < 8; ++j) acc[a][t][j] *= alp[j];
    }

    // A fragments (two 16-row tiles) from LDS probabilities
    Frag af[2];
#pragma unroll
    for (int a = 0; a < 2; ++a) {
      const int rowb = (a * 16 + sub) * 17;
#pragma unroll
      for (int r = 0; r < 4; ++r) af[a].i[r] = sm_p[rowb + r + 4 * half];
#pragma unroll
      for (int r = 4; r < 8; ++r) af[a].i[r] = sm_p[rowb + 8 + (r - 4) + 4 * half];
    }

    // 8 WMMAs: each B fragment feeds both A tiles
#pragma unroll
    for (int t = 0; t < 4; ++t) {
      acc[0][t] = __builtin_amdgcn_wmma_f32_16x16x32_bf16(
          false, af[0].h, false, bf[t].h, (short)0, acc[0][t], false, false);
      acc[1][t] = __builtin_amdgcn_wmma_f32_16x16x32_bf16(
          false, af[1].h, false, bf[t].h, (short)0, acc[1][t], false, false);
    }
  }

  // epilogue: 1/l normalization, sigmoid gate, bf16 store
  sm_s[lane] = 1.0f / l;
  __syncthreads();
#pragma unroll
  for (int a = 0; a < 2; ++a) {
    float inv[8];
#pragma unroll
    for (int j = 0; j < 8; ++j) inv[j] = sm_s[a * 16 + j + 8 * half];
#pragma unroll
    for (int t = 0; t < 4; ++t)
#pragma unroll
      for (int j = 0; j < 8; ++j) {
        int q   = q0 + a * 16 + j + 8 * half;
        int col = h * 64 + t * 16 + sub;
        size_t idx = ((size_t)b * Qq + q) * HD + col;
        og[idx] = f2bf(acc[a][t][j] * inv[j] * g[idx]);
      }
  }
}

// ---------------------------------------------------------------------------
// out = og @ W_o   (bf16 x bf16 -> fp32)
// grid: (BQ/16, CV/64), block: 32
// ---------------------------------------------------------------------------
__global__ void k_out(const unsigned short* __restrict__ og,
                      const unsigned short* __restrict__ WoT,
                      float* __restrict__ out) {
  const int mt = blockIdx.x, nt = blockIdx.y;
  const int lane = threadIdx.x & 31, sub = lane & 15, half = lane >> 4;
  v8f acc[4];
#pragma unroll
  for (int t = 0; t < 4; ++t)
#pragma unroll
    for (int j = 0; j < 8; ++j) acc[t][j] = 0.0f;

  const int* A  = (const int*)og  + (size_t)(mt * 16) * (HD / 2);
  const int* Bt = (const int*)WoT + (size_t)(nt * 64) * (HD / 2);
  gemm_tile(A, HD / 2, Bt, HD / 2, HD / 2, acc);

#pragma unroll
  for (int t = 0; t < 4; ++t)
#pragma unroll
    for (int j = 0; j < 8; ++j) {
      int row = mt * 16 + j + 8 * half;
      int col = nt * 64 + t * 16 + sub;
      out[(size_t)row * CVc + col] = acc[t][j];
    }
}

// ---------------------------------------------------------------------------
extern "C" void kernel_launch(void* const* d_in, const int* in_sizes, int n_in,
                              void* d_out, int out_size, void* d_ws, size_t ws_size,
                              hipStream_t stream) {
  const float* x     = (const float*)d_in[0];
  const float* bias1 = (const float*)d_in[1];
  const float* bias2 = (const float*)d_in[2];
  const float* W_v   = (const float*)d_in[3];
  const float* W_g   = (const float*)d_in[4];
  const float* W_o   = (const float*)d_in[5];
  float* out = (float*)d_out;

  char* ws = (char*)d_ws;
  unsigned short* xb  = (unsigned short*)(ws);                         // 4 MB
  unsigned short* WvT = (unsigned short*)(ws + 4194304);               // 512 KB
  unsigned short* WgT = (unsigned short*)(ws + 4718592);               // 512 KB
  unsigned short* WoT = (unsigned short*)(ws + 5242880);               // 512 KB
  unsigned short* vT  = (unsigned short*)(ws + 5767168);               // 4 MB
  float*          g   = (float*)        (ws + 9961472);                // 8 MB
  unsigned short* og  = (unsigned short*)(ws + 18350080);              // 4 MB
  // total workspace: ~21.5 MB

  // 1) precision conversion
  k_cvt_x<<<dim3((BQ * CVc + 255) / 256), dim3(256), 0, stream>>>(x, xb, BQ * CVc);
  k_cvt_wT<<<dim3(2, 512), dim3(256), 0, stream>>>(W_v, WvT);
  k_cvt_wT<<<dim3(2, 512), dim3(256), 0, stream>>>(W_g, WgT);
  k_cvt_wT<<<dim3(2, 512), dim3(256), 0, stream>>>(W_o, WoT);

  // 2) v (transposed bf16) and gate (fp32)
  k_vg<<<dim3(BQ / 16, HD / 64, 2), dim3(32), 0, stream>>>(xb, WvT, WgT, vT, g);

  // 3) fused bias-softmax attention + gating (single pass over 268MB bias1)
  k_attn<<<dim3(Qq / 32, Hh, Bb), dim3(32), 0, stream>>>(bias1, bias2, vT, g, og);

  // 4) output projection
  k_out<<<dim3(BQ / 16, CVc / 64), dim3(32), 0, stream>>>(og, WoT, out);
}
